// GCNClassifier_37606733644271
// MI455X (gfx1250) — matrix operand order repaired and checked by reading, not compiled, not run on hardware
//
#include <hip/hip_runtime.h>
#include <hip/hip_bf16.h>

// ---------------------------------------------------------------------------
// GCN (3x GCNConv + mean-pool + linear) for gfx1250 (MI455X, wave32, WMMA).
// GEMMs: V_WMMA_F32_16X16X32_BF16 with bf16x3 compensation (hi*hi + hi*lo +
// lo*hi, f32 accumulate) for near-fp32 accuracy.
//   - W pre-packed once into fragment-layout hi/lo pairs (no per-tile conv)
//   - A strip converted once per block into LDS fragments (ds_load_b128)
// Aggregation: per-edge wave scatter with native f32 atomics (51MB activation
// matrix is L2-resident on the 192MB L2).
// ---------------------------------------------------------------------------

#define NNODES   100000
#define NEDGES   1600000
#define FDIM     128
#define NGRAPHS  64
#define WFRAG    8192   // uint32 packed pairs per 128x128 weight fragment set

typedef __attribute__((ext_vector_type(16))) __bf16        v16bf;
typedef __attribute__((ext_vector_type(8)))  unsigned int  v8u;
typedef __attribute__((ext_vector_type(8)))  float         v8f;

// round-to-nearest-even f32 -> bf16 bits, returned in high 16 of a u32
static __device__ __forceinline__ unsigned int bf16_rne_u32(float f) {
    unsigned int u = __float_as_uint(f);
    return u + 0x7FFFu + ((u >> 16) & 1u);
}
static __device__ __forceinline__ float bf16hi_to_f32(unsigned int r) {
    return __uint_as_float(r & 0xFFFF0000u);
}
// pack two bf16 (from floats a0,a1) into one u32: [15:0]=bf16(a0) [31:16]=bf16(a1)
static __device__ __forceinline__ unsigned int pack_bf16x2(float a0, float a1) {
    return (bf16_rne_u32(a0) >> 16) | (bf16_rne_u32(a1) & 0xFFFF0000u);
}

static __device__ __forceinline__ float atomic_add_f32(float* p, float v) {
    return __hip_atomic_fetch_add(p, v, __ATOMIC_RELAXED, __HIP_MEMORY_SCOPE_AGENT);
}

// ---------------------------------------------------------------------------
// Pre-pack one 128x128 f32 weight into B-fragment layout hi/lo bf16 pairs.
// Element e = ((kc*8 + ct)*32 + lane)*8 + i  holds K rows {krow, krow+1},
// column ct*16 + (lane&15), where krow = kc*32 + 2i + (lane>>4)*16.
// ---------------------------------------------------------------------------
__global__ void pack_weight_kernel(const float* __restrict__ W,
                                   unsigned int* __restrict__ hi,
                                   unsigned int* __restrict__ lo) {
    int e = blockIdx.x * blockDim.x + threadIdx.x;
    if (e >= WFRAG) return;
    const int i    = e & 7;
    const int lane = (e >> 3) & 31;
    const int ct   = (e >> 8) & 7;
    const int kc   = e >> 11;
    const int m    = lane & 15;
    const int kh   = lane >> 4;
    const int krow = kc * 32 + 2 * i + kh * 16;
    const int col  = ct * 16 + m;
    const float b0 = W[(size_t)krow       * FDIM + col];
    const float b1 = W[(size_t)(krow + 1) * FDIM + col];
    const unsigned int r0 = bf16_rne_u32(b0);
    const unsigned int r1 = bf16_rne_u32(b1);
    hi[e] = (r0 >> 16) | (r1 & 0xFFFF0000u);
    lo[e] = pack_bf16x2(b0 - bf16hi_to_f32(r0), b1 - bf16hi_to_f32(r1));
}

// ---------------------------------------------------------------------------
// GEMM: Out[N,128] = act(A)[N,128] @ W[128,128], act = ReLU if RELU.
// Block = 256 threads = 8 waves; block owns a 16-row strip; wave w owns
// column tile w. A fragments staged once in LDS; B fragments pre-packed.
// ---------------------------------------------------------------------------
template <int RELU>
__global__ __launch_bounds__(256) void gemm_bf16x3_kernel(
    const float* __restrict__ A,
    const unsigned int* __restrict__ Whi,
    const unsigned int* __restrict__ Wlo,
    float* __restrict__ Out, int nrows)
{
    __shared__ unsigned int sAhi[4 * 32 * 8];   // [kc][lane][i] packed pairs
    __shared__ unsigned int sAlo[4 * 32 * 8];

    const int r0 = blockIdx.x * 16;
    if (r0 >= nrows) return;                     // uniform per block

    // ---- stage A strip (16x128) into hi/lo fragments: each element once ---
    #pragma unroll
    for (int r = 0; r < 4; ++r) {
        const int e    = threadIdx.x + r * 256;  // [0,1024)
        const int i    = e & 7;
        const int lane = (e >> 3) & 31;
        const int kc   = e >> 8;
        const int m    = lane & 15;
        const int kh   = lane >> 4;
        const int k0   = 2 * i + kh * 16;
        const float2 a = *(const float2*)(A + (size_t)(r0 + m) * FDIM + kc * 32 + k0);
        float a0 = a.x, a1 = a.y;
        if (RELU) { a0 = fmaxf(a0, 0.0f); a1 = fmaxf(a1, 0.0f); }
        const unsigned int u0 = bf16_rne_u32(a0);
        const unsigned int u1 = bf16_rne_u32(a1);
        sAhi[e] = (u0 >> 16) | (u1 & 0xFFFF0000u);
        sAlo[e] = pack_bf16x2(a0 - bf16hi_to_f32(u0), a1 - bf16hi_to_f32(u1));
    }
    __syncthreads();

    const int wave = threadIdx.x >> 5;           // column tile
    const int lane = threadIdx.x & 31;
    const int m    = lane & 15;
    const int kh   = lane >> 4;

    v8f acc = {};
    #pragma unroll
    for (int kc = 0; kc < 4; ++kc) {
        const int abase = (kc * 32 + lane) * 8;
        const uint4 ah0 = *(const uint4*)(sAhi + abase);
        const uint4 ah1 = *(const uint4*)(sAhi + abase + 4);
        const uint4 al0 = *(const uint4*)(sAlo + abase);
        const uint4 al1 = *(const uint4*)(sAlo + abase + 4);
        const size_t bbase = (size_t)(((kc * 8 + wave) * 32 + lane) * 8);
        const uint4 bh0 = *(const uint4*)(Whi + bbase);
        const uint4 bh1 = *(const uint4*)(Whi + bbase + 4);
        const uint4 bl0 = *(const uint4*)(Wlo + bbase);
        const uint4 bl1 = *(const uint4*)(Wlo + bbase + 4);

        const v8u ahu = {ah0.x, ah0.y, ah0.z, ah0.w, ah1.x, ah1.y, ah1.z, ah1.w};
        const v8u alu = {al0.x, al0.y, al0.z, al0.w, al1.x, al1.y, al1.z, al1.w};
        const v8u bhu = {bh0.x, bh0.y, bh0.z, bh0.w, bh1.x, bh1.y, bh1.z, bh1.w};
        const v8u blu = {bl0.x, bl0.y, bl0.z, bl0.w, bl1.x, bl1.y, bl1.z, bl1.w};
        const v16bf ahi = __builtin_bit_cast(v16bf, ahu);
        const v16bf alo = __builtin_bit_cast(v16bf, alu);
        const v16bf bhi = __builtin_bit_cast(v16bf, bhu);
        const v16bf blo = __builtin_bit_cast(v16bf, blu);

        acc = __builtin_amdgcn_wmma_f32_16x16x32_bf16(false, alo, false, bhi,
                                                      (short)0, acc, false, false);
        acc = __builtin_amdgcn_wmma_f32_16x16x32_bf16(false, ahi, false, blo,
                                                      (short)0, acc, false, false);
        acc = __builtin_amdgcn_wmma_f32_16x16x32_bf16(false, ahi, false, bhi,
                                                      (short)0, acc, false, false);
    }

    // C/D layout: VGPR i -> M = i (lanes 0-15) / M = i+8 (lanes 16-31)
    #pragma unroll
    for (int i = 0; i < 8; ++i) {
        const int row = r0 + i + kh * 8;
        Out[(size_t)row * FDIM + wave * 16 + m] = acc[i];
    }
}

// --- degree / normalization -------------------------------------------------
__global__ void deg_init_kernel(float* deg, int n) {
    int i = blockIdx.x * blockDim.x + threadIdx.x;
    if (i < n) deg[i] = 1.0f;                       // self-loop contribution
}
__global__ void deg_edges_kernel(const int* __restrict__ dst, float* deg, int ne) {
    int e = blockIdx.x * blockDim.x + threadIdx.x;
    if (e < ne) atomic_add_f32(&deg[dst[e]], 1.0f);
}
__global__ void deg_to_dis_kernel(float* deg, int n) {
    int i = blockIdx.x * blockDim.x + threadIdx.x;
    if (i < n) deg[i] = rsqrtf(deg[i]);             // deg >= 1 always
}

// --- init aggregation buffer to broadcast bias ------------------------------
__global__ void bias_init_kernel(float* __restrict__ agg,
                                 const float* __restrict__ b, int total) {
    int idx = blockIdx.x * blockDim.x + threadIdx.x;
    if (idx < total) agg[idx] = b[idx & (FDIM - 1)];
}

// --- per-edge scatter: one wave per (edge or self-loop) ---------------------
__global__ __launch_bounds__(256) void scatter_kernel(
    const float* __restrict__ T, const int* __restrict__ src,
    const int* __restrict__ dst, const float* __restrict__ dis,
    float* __restrict__ agg, int ne, int nn)
{
    const int wave = blockIdx.x * (blockDim.x >> 5) + (threadIdx.x >> 5);
    const int lane = threadIdx.x & 31;
    if (wave >= ne + nn) return;
    int s, d;
    if (wave < ne) { s = src[wave]; d = dst[wave]; }
    else           { s = wave - ne; d = s; }        // self-loop
    const float nrm = dis[s] * dis[d];
    const float4 v = ((const float4*)(T + (size_t)s * FDIM))[lane];
    float* o = agg + (size_t)d * FDIM + lane * 4;
    atomic_add_f32(o + 0, v.x * nrm);
    atomic_add_f32(o + 1, v.y * nrm);
    atomic_add_f32(o + 2, v.z * nrm);
    atomic_add_f32(o + 3, v.w * nrm);
}

// --- mean pool --------------------------------------------------------------
__global__ void pool_zero_kernel(float* sums, float* cnts) {
    int i = blockIdx.x * blockDim.x + threadIdx.x;
    if (i < NGRAPHS * FDIM) sums[i] = 0.0f;
    if (i < NGRAPHS) cnts[i] = 0.0f;
}
__global__ __launch_bounds__(256) void pool_scatter_kernel(
    const float* __restrict__ H, const int* __restrict__ batch,
    float* __restrict__ sums, float* __restrict__ cnts, int n)
{
    const int node = blockIdx.x * (blockDim.x >> 5) + (threadIdx.x >> 5);
    const int lane = threadIdx.x & 31;
    if (node >= n) return;
    const int g = batch[node];
    const float4 v = ((const float4*)(H + (size_t)node * FDIM))[lane];
    float* o = sums + (size_t)g * FDIM + lane * 4;
    atomic_add_f32(o + 0, v.x);
    atomic_add_f32(o + 1, v.y);
    atomic_add_f32(o + 2, v.z);
    atomic_add_f32(o + 3, v.w);
    if (lane == 0) atomic_add_f32(&cnts[g], 1.0f);
}
__global__ void pool_final_kernel(const float* __restrict__ sums,
                                  const float* __restrict__ cnts,
                                  float* __restrict__ pooled) {
    int i = blockIdx.x * blockDim.x + threadIdx.x;
    if (i < NGRAPHS * FDIM) pooled[i] = sums[i] / fmaxf(cnts[i >> 7], 1.0f);
}
__global__ void final_linear_kernel(const float* __restrict__ pooled,
                                    const float* __restrict__ Wlin,
                                    const float* __restrict__ blin,
                                    float* __restrict__ out) {
    int t = blockIdx.x * blockDim.x + threadIdx.x;   // 128 = 64 graphs x 2
    if (t >= NGRAPHS * 2) return;
    const int g = t >> 1, c = t & 1;
    float s = blin[c];
    #pragma unroll 8
    for (int k = 0; k < FDIM; ++k) s += pooled[g * FDIM + k] * Wlin[k * 2 + c];
    out[t] = s;
}

// ---------------------------------------------------------------------------
extern "C" void kernel_launch(void* const* d_in, const int* in_sizes, int n_in,
                              void* d_out, int out_size, void* d_ws, size_t ws_size,
                              hipStream_t stream) {
    (void)in_sizes; (void)n_in; (void)out_size; (void)ws_size;
    const float* x    = (const float*)d_in[0];
    const float* W1   = (const float*)d_in[1];
    const float* b1   = (const float*)d_in[2];
    const float* W2   = (const float*)d_in[3];
    const float* b2   = (const float*)d_in[4];
    const float* W3   = (const float*)d_in[5];
    const float* b3   = (const float*)d_in[6];
    const float* Wlin = (const float*)d_in[7];
    const float* blin = (const float*)d_in[8];
    const int*   src  = (const int*)d_in[9];          // edge_index[0]
    const int*   dst  = src + NEDGES;                 // edge_index[1]
    const int*   batch = (const int*)d_in[10];

    float* out    = (float*)d_out;                    // [64*2] then pooled [64*128]
    float* pooled = out + NGRAPHS * 2;

    float* bufA = (float*)d_ws;                       // GEMM output (messages)
    float* bufB = bufA + (size_t)NNODES * FDIM;       // aggregation / activations
    float* dis  = bufB + (size_t)NNODES * FDIM;       // degree -> 1/sqrt(deg)
    float* sums = dis + NNODES;
    float* cnts = sums + NGRAPHS * FDIM;
    unsigned int* wpk = (unsigned int*)(cnts + NGRAPHS);  // 6 x WFRAG u32
    unsigned int* W1hi = wpk + 0 * WFRAG;
    unsigned int* W1lo = wpk + 1 * WFRAG;
    unsigned int* W2hi = wpk + 2 * WFRAG;
    unsigned int* W2lo = wpk + 3 * WFRAG;
    unsigned int* W3hi = wpk + 4 * WFRAG;
    unsigned int* W3lo = wpk + 5 * WFRAG;

    const int packBlocks = WFRAG / 256;
    const int waveBlocks = ((NEDGES + NNODES) + 7) / 8;  // 8 waves / 256-thr block

    // weight fragment packing + normalization
    pack_weight_kernel<<<packBlocks, 256, 0, stream>>>(W1, W1hi, W1lo);
    pack_weight_kernel<<<packBlocks, 256, 0, stream>>>(W2, W2hi, W2lo);
    pack_weight_kernel<<<packBlocks, 256, 0, stream>>>(W3, W3hi, W3lo);
    deg_init_kernel<<<(NNODES + 255) / 256, 256, 0, stream>>>(dis, NNODES);
    deg_edges_kernel<<<(NEDGES + 255) / 256, 256, 0, stream>>>(dst, dis, NEDGES);
    deg_to_dis_kernel<<<(NNODES + 255) / 256, 256, 0, stream>>>(dis, NNODES);

    // layer 1: T = x @ W1 ; agg = b1 + scatter(norm * T)
    gemm_bf16x3_kernel<0><<<NNODES / 16, 256, 0, stream>>>(x, W1hi, W1lo, bufA, NNODES);
    bias_init_kernel<<<(NNODES * FDIM + 255) / 256, 256, 0, stream>>>(bufB, b1, NNODES * FDIM);
    scatter_kernel<<<waveBlocks, 256, 0, stream>>>(bufA, src, dst, dis, bufB, NEDGES, NNODES);

    // layer 2 (ReLU folded into A load)
    gemm_bf16x3_kernel<1><<<NNODES / 16, 256, 0, stream>>>(bufB, W2hi, W2lo, bufA, NNODES);
    bias_init_kernel<<<(NNODES * FDIM + 255) / 256, 256, 0, stream>>>(bufB, b2, NNODES * FDIM);
    scatter_kernel<<<waveBlocks, 256, 0, stream>>>(bufA, src, dst, dis, bufB, NEDGES, NNODES);

    // layer 3 (no ReLU on output)
    gemm_bf16x3_kernel<1><<<NNODES / 16, 256, 0, stream>>>(bufB, W3hi, W3lo, bufA, NNODES);
    bias_init_kernel<<<(NNODES * FDIM + 255) / 256, 256, 0, stream>>>(bufB, b3, NNODES * FDIM);
    scatter_kernel<<<waveBlocks, 256, 0, stream>>>(bufA, src, dst, dis, bufB, NEDGES, NNODES);

    // mean pool + classifier head
    pool_zero_kernel<<<(NGRAPHS * FDIM + 255) / 256, 256, 0, stream>>>(sums, cnts);
    pool_scatter_kernel<<<(NNODES + 7) / 8, 256, 0, stream>>>(bufB, batch, sums, cnts, NNODES);
    pool_final_kernel<<<(NGRAPHS * FDIM + 255) / 256, 256, 0, stream>>>(sums, cnts, pooled);
    final_linear_kernel<<<1, 128, 0, stream>>>(pooled, Wlin, blin, out);
}
